// SA_Layer_23536420782435
// MI455X (gfx1250) — matrix operand order repaired and checked by reading, not compiled, not run on hardware
//
#include <hip/hip_runtime.h>

// Problem constants (from reference): B=4, C=128, N=4096, D=128
#define B_ 4
#define C_ 128
#define N_ 4096
#define D_ 128
#define MT_ 64            // attention output columns (m) per block
#define LDQ_ 136          // padded LDS row stride for 128-wide bf16 tiles (68 dwords)
#define LDP_ 72           // sizes each P buffer (128 * LDP_ bf16); P is stored [m][n] with LDQ_ rows
#define LDO_ 68           // padded f32 row stride for the x_r transpose buffer
#define BN_EPS 1e-5f

typedef __attribute__((ext_vector_type(16))) __bf16 v16bf;
typedef __attribute__((ext_vector_type(8)))  float  v8f;
typedef __attribute__((ext_vector_type(4)))  int    v4i;

static __device__ __forceinline__ v8f wmma_bf16(v16bf a, v16bf b, v8f c) {
  return __builtin_amdgcn_wmma_f32_16x16x32_bf16(
      false, a, false, b, (short)0, c, false, false);
}

// 32-byte bf16 vector load with only 16B alignment (padded LDS rows)
static __device__ __forceinline__ v16bf load_v16bf16(const __bf16* p) {
  union { v16bf v; uint4 u[2]; } t;
  t.u[0] = *(const uint4*)(p);
  t.u[1] = *(const uint4*)(p + 8);
  return t.v;
}

// ---- CDNA5 async global<->LDS copies ---------------------------------------
#if defined(__has_builtin)
#  if __has_builtin(__builtin_amdgcn_global_load_async_to_lds_b128)
#    define USE_ASYNC_COPY 1
#  endif
#endif

static __device__ __forceinline__ void copy16_g2l(const void* g, void* l) {
#ifdef USE_ASYNC_COPY
  __builtin_amdgcn_global_load_async_to_lds_b128((v4i*)g, (v4i*)l, 0, 0);
#else
  *(uint4*)l = *(const uint4*)g;
#endif
}
static __device__ __forceinline__ void copy16_l2g(void* g, const void* l) {
#if __has_builtin(__builtin_amdgcn_global_store_async_from_lds_b128)
  __builtin_amdgcn_global_store_async_from_lds_b128((v4i*)g, (v4i*)l, 0, 0);
#else
  *(uint4*)g = *(const uint4*)l;
#endif
}
static __device__ __forceinline__ void async_wait0() {
#ifdef USE_ASYNC_COPY
#  if __has_builtin(__builtin_amdgcn_s_wait_asynccnt)
  __builtin_amdgcn_s_wait_asynccnt(0);
#  else
  asm volatile("s_wait_asynccnt 0" ::: "memory");
#  endif
#endif
}
static __device__ __forceinline__ void async_wait2() {  // allow newest stage in flight
#ifdef USE_ASYNC_COPY
#  if __has_builtin(__builtin_amdgcn_s_wait_asynccnt)
  __builtin_amdgcn_s_wait_asynccnt(2);
#  else
  asm volatile("s_wait_asynccnt 2" ::: "memory");
#  endif
#endif
}

// ---------------------------------------------------------------------------
// Kernel 0: one-time f32 -> bf16 conversion of the projection weights.
// ---------------------------------------------------------------------------
__global__ __launch_bounds__(256)
void convw_kernel(const float* __restrict__ Wqk, const float* __restrict__ Wv,
                  __bf16* __restrict__ wqb, __bf16* __restrict__ wvb) {
  int i = blockIdx.x * 256 + threadIdx.x;        // 0 .. 16383
  wqb[i] = (__bf16)Wqk[i];
  wvb[i] = (__bf16)Wv[i];
}

// ---------------------------------------------------------------------------
// Kernel 1: WMMA projections over 128-point tiles.
//   q[b][n][d] = Wqk[d,:]·x[b,:,n]   (bf16, n-major)  -> packed b128 stores
//   v[b][d][n] = Wv[d,:]·x[b,:,n]+bv (bf16, d-major)  -> LDS-transposed stores
// ---------------------------------------------------------------------------
__global__ __launch_bounds__(256)
void proj_kernel(const float* __restrict__ x,
                 const __bf16* __restrict__ wqb, const __bf16* __restrict__ wvb,
                 const float* __restrict__ bv,
                 __bf16* __restrict__ qb, __bf16* __restrict__ vb) {
  const int wave = threadIdx.x >> 5, lane = threadIdx.x & 31;
  const int lrow = lane & 15, lhi = lane >> 4;
  const int b  = blockIdx.x / (N_ / 128);
  const int n0 = (blockIdx.x % (N_ / 128)) * 128;

  __shared__ __align__(16) __bf16 sXT[128 * LDQ_];   // x^T tile [n][c], padded
  __shared__ __align__(16) __bf16 sVT[8][16 * 24];   // per-wave v-tile transpose

  // stage + convert + transpose: 128c x 128n floats, float4 granular
  for (int i = threadIdx.x; i < 128 * 32; i += 256) {
    int c = i >> 5, seg = i & 31;
    float4 xv = *(const float4*)(x + ((size_t)b * C_ + c) * N_ + n0 + seg * 4);
    int nb = seg * 4;
    sXT[(nb + 0) * LDQ_ + c] = (__bf16)xv.x;
    sXT[(nb + 1) * LDQ_ + c] = (__bf16)xv.y;
    sXT[(nb + 2) * LDQ_ + c] = (__bf16)xv.z;
    sXT[(nb + 3) * LDQ_ + c] = (__bf16)xv.w;
  }
  __syncthreads();

  const int dt = wave * 16;
  v16bf AQ[4], AV[4];
  #pragma unroll
  for (int k = 0; k < 4; ++k) {
    AQ[k] = *(const v16bf*)(wqb + (size_t)(dt + lrow) * C_ + k * 32 + lhi * 16);
    AV[k] = *(const v16bf*)(wvb + (size_t)(dt + lrow) * C_ + k * 32 + lhi * 16);
  }
  float bvv[8];
  #pragma unroll
  for (int r = 0; r < 8; ++r) bvv[r] = bv[dt + r + 8 * lhi];

  #pragma unroll 1
  for (int nt = 0; nt < 8; ++nt) {
    v16bf Bx[4];
    #pragma unroll
    for (int k = 0; k < 4; ++k)
      Bx[k] = load_v16bf16(&sXT[(nt * 16 + lrow) * LDQ_ + k * 32 + lhi * 16]);
    v8f oq = {}, ov = {};
    #pragma unroll
    for (int k = 0; k < 4; ++k) {
      oq = wmma_bf16(AQ[k], Bx[k], oq);
      ov = wmma_bf16(AV[k], Bx[k], ov);
    }
    // q: this lane's 8 values are consecutive in d -> one b128 store
    {
      union { __bf16 h[8]; uint4 u; } pk;
      #pragma unroll
      for (int r = 0; r < 8; ++r) pk.h[r] = (__bf16)oq[r];
      int n = n0 + nt * 16 + lrow;
      *(uint4*)(qb + ((size_t)b * N_ + n) * D_ + dt + 8 * lhi) = pk.u;
    }
    // v: transpose through wave-private LDS, then 16B row-segment stores
    {
      #pragma unroll
      for (int r = 0; r < 8; ++r)
        sVT[wave][(r + 8 * lhi) * 24 + lrow] = (__bf16)(ov[r] + bvv[r]);
      int row = lane >> 1, offn = (lane & 1) * 8;     // LDS in-order per wave
      uint4 vv = *(uint4*)&sVT[wave][row * 24 + offn];
      *(uint4*)(vb + ((size_t)b * D_ + dt + row) * N_ + n0 + nt * 16 + offn) = vv;
    }
  }
}

// ---------------------------------------------------------------------------
// Kernel 2: flash pass 1 — per-row (n) safe-softmax stats of energy = Q·Qᵀ.
// 32-column m-tiles, 3 LDS buffers, depth-2 async pipeline.
// ---------------------------------------------------------------------------
__global__ __launch_bounds__(256)
void rowstats_kernel(const __bf16* __restrict__ qb,
                     float* __restrict__ rowmax, float* __restrict__ rowsum) {
  const int wave = threadIdx.x >> 5, lane = threadIdx.x & 31;
  const int lrow = lane & 15, lhi = lane >> 4;
  const int b     = blockIdx.x / (N_ / 128);
  const int strip = blockIdx.x % (N_ / 128);
  const int n0 = strip * 128 + wave * 16;

  __shared__ __align__(16) __bf16 sB[3][32 * LDQ_];   // 3 x ~8.5 KB m-tiles

  const __bf16* qbase = qb + (size_t)b * N_ * D_;

  v16bf A[4];
  #pragma unroll
  for (int k = 0; k < 4; ++k)
    A[k] = *(const v16bf*)(qbase + (size_t)(n0 + lrow) * D_ + k * 32 + lhi * 16);

  const int NT = N_ / 32;
  #define STAGE_TILE(tile, buf)                                                \
    {                                                                          \
      _Pragma("unroll")                                                        \
      for (int j = 0; j < 2; ++j) {                                            \
        int i = threadIdx.x + j * 256;                                         \
        int row = i >> 4, seg = i & 15;                                        \
        copy16_g2l(qbase + (size_t)((tile) * 32 + row) * D_ + seg * 8,         \
                   &sB[buf][row * LDQ_ + seg * 8]);                            \
      }                                                                        \
    }
  STAGE_TILE(0, 0)
  STAGE_TILE(1, 1)
  async_wait2();       // tile 0 landed, tile 1 may still fly
  __syncthreads();

  float M[8], S[8];
  #pragma unroll
  for (int r = 0; r < 8; ++r) { M[r] = -1e30f; S[r] = 0.f; }

  for (int t = 0; t < NT; ++t) {
    const int buf = t % 3;
    if (t + 2 < NT) STAGE_TILE(t + 2, (t + 2) % 3)
    v16bf B0[4], B1[4];
    #pragma unroll
    for (int k = 0; k < 4; ++k) {
      B0[k] = load_v16bf16(&sB[buf][lrow * LDQ_ + k * 32 + lhi * 16]);
      B1[k] = load_v16bf16(&sB[buf][(16 + lrow) * LDQ_ + k * 32 + lhi * 16]);
    }
    v8f e0 = {}, e1 = {};
    #pragma unroll
    for (int k = 0; k < 4; ++k) {     // interleaved: no RAW between neighbors
      e0 = wmma_bf16(A[k], B0[k], e0);
      e1 = wmma_bf16(A[k], B1[k], e1);
    }
    #pragma unroll
    for (int r = 0; r < 8; ++r) {
      float nm = fmaxf(M[r], e0[r]);
      S[r] = S[r] * __expf(M[r] - nm) + __expf(e0[r] - nm);
      M[r] = nm;
      nm = fmaxf(M[r], e1[r]);
      S[r] = S[r] * __expf(M[r] - nm) + __expf(e1[r] - nm);
      M[r] = nm;
    }
    if (t + 2 < NT) async_wait2(); else async_wait0();
    __syncthreads();
  }
  #undef STAGE_TILE

  #pragma unroll
  for (int r = 0; r < 8; ++r) {
    #pragma unroll
    for (int i = 0; i < 4; ++i) {
      int off = 1 << i;
      float Mo = __shfl_xor(M[r], off, 32);
      float So = __shfl_xor(S[r], off, 32);
      float nm = fmaxf(M[r], Mo);
      S[r] = S[r] * __expf(M[r] - nm) + So * __expf(Mo - nm);
      M[r] = nm;
    }
  }
  if (lrow == 0) {
    #pragma unroll
    for (int r = 0; r < 8; ++r) {
      int row = n0 + r + 8 * lhi;
      rowmax[(size_t)b * N_ + row] = M[r];
      rowsum[(size_t)b * N_ + row] = S[r];
    }
  }
}

// ---------------------------------------------------------------------------
// Kernel 3: flash pass 2, 64 m-columns per block (d-split mapping).
// k-outer e-phase, batch-preloaded LDS operands, padded rows; x_r written via
// an in-LDS transpose + coalesced (async) b128 stores.
// ---------------------------------------------------------------------------
__global__ __launch_bounds__(256)
void attend_kernel(const __bf16* __restrict__ qb, const __bf16* __restrict__ vb,
                   const float* __restrict__ rowmax, const float* __restrict__ rowsum,
                   float* xr) {
  const int wave = threadIdx.x >> 5, lane = threadIdx.x & 31;
  const int lrow = lane & 15, lhi = lane >> 4;
  const int b  = blockIdx.x / (N_ / MT_);
  const int m0 = (blockIdx.x % (N_ / MT_)) * MT_;

  __shared__ __align__(16) __bf16 sBq[MT_ * LDQ_];           // ~17 KB
  __shared__ __align__(16) char   sPraw[2 * 128 * LDP_ * 2]; // 36 KB (P / x_r overlay)
  __shared__ float sCol[8][MT_];
  __shared__ float sTot[MT_];
  __bf16* sPb = (__bf16*)sPraw;

  const __bf16* qbase = qb + (size_t)b * N_ * D_;
  const __bf16* vbase = vb + (size_t)b * D_ * N_;
  const float*  rmb   = rowmax + (size_t)b * N_;
  const float*  rsb   = rowsum + (size_t)b * N_;

  for (int i = threadIdx.x; i < MT_ * 16; i += 256) {
    int row = i >> 4, seg = i & 15;
    copy16_g2l(qbase + (size_t)(m0 + row) * D_ + seg * 8,
               &sBq[row * LDQ_ + seg * 8]);
  }
  async_wait0();
  __syncthreads();

  const int d0 = wave * 16;
  v8f   O[4] = {};
  float scol[4] = {0.f, 0.f, 0.f, 0.f};

  v16bf Aq[4];
  #pragma unroll
  for (int k = 0; k < 4; ++k)
    Aq[k] = *(const v16bf*)(qbase + (size_t)(wave * 16 + lrow) * D_ + k * 32 + lhi * 16);

  #pragma unroll 1
  for (int sc = 0; sc < N_ / 128; ++sc) {
    const int buf = sc & 1;
    const int n0  = sc * 128 + wave * 16;

    float rmv[8], rsv[8];
    #pragma unroll
    for (int r = 0; r < 8; ++r) {
      int row = n0 + r + 8 * lhi;
      rmv[r] = rmb[row];
      rsv[r] = 1.f / rsb[row];
    }
    v8f eacc[4] = {};
    #pragma unroll
    for (int k = 0; k < 4; ++k) {
      v16bf Bk[4];
      #pragma unroll
      for (int mt = 0; mt < 4; ++mt)
        Bk[mt] = load_v16bf16(&sBq[(mt * 16 + lrow) * LDQ_ + k * 32 + lhi * 16]);
      #pragma unroll
      for (int mt = 0; mt < 4; ++mt)
        eacc[mt] = wmma_bf16(Aq[k], Bk[mt], eacc[mt]);
    }
    #pragma unroll
    for (int mt = 0; mt < 4; ++mt) {
      #pragma unroll
      for (int r = 0; r < 8; ++r) {
        float p = __expf(eacc[mt][r] - rmv[r]) * rsv[r];
        scol[mt] += p;
        sPb[buf * (128 * LDP_) + (mt * 16 + lrow) * LDQ_ + wave * 16 + r + 8 * lhi] =
            (__bf16)p;
      }
    }
    v16bf An[4];
    if (sc + 1 < N_ / 128) {
      #pragma unroll
      for (int k = 0; k < 4; ++k)
        An[k] = *(const v16bf*)(qbase +
                 (size_t)((sc + 1) * 128 + wave * 16 + lrow) * D_ + k * 32 + lhi * 16);
    }
    __syncthreads();
    #pragma unroll
    for (int c = 0; c < 4; ++c) {
      v16bf Av = *(const v16bf*)(vbase + (size_t)(d0 + lrow) * N_ +
                                 sc * 128 + c * 32 + lhi * 16);
      v16bf PBt[4];
      #pragma unroll
      for (int mt = 0; mt < 4; ++mt)
        PBt[mt] = load_v16bf16(&sPb[buf * (128 * LDP_) + (mt * 16 + lrow) * LDQ_ + c * 32 + lhi * 16]);
      #pragma unroll
      for (int mt = 0; mt < 4; ++mt)
        O[mt] = wmma_bf16(Av, PBt[mt], O[mt]);
    }
    if (sc + 1 < N_ / 128) {
      #pragma unroll
      for (int k = 0; k < 4; ++k) Aq[k] = An[k];
    }
  }

  // ---- column sums: fold lane halves, then across waves
  #pragma unroll
  for (int mt = 0; mt < 4; ++mt) {
    float s = scol[mt] + __shfl_xor(scol[mt], 16, 32);
    if (lane < 16) sCol[wave][mt * 16 + lane] = s;
  }
  __syncthreads();
  if (threadIdx.x < MT_) {
    float tot = 0.f;
    #pragma unroll
    for (int w2 = 0; w2 < 8; ++w2) tot += sCol[w2][threadIdx.x];
    sTot[threadIdx.x] = 1.f / (1e-9f + tot);
  }
  __syncthreads();

  // ---- normalize, transpose through LDS (overlaying sP), coalesced stores
  float* sPf = (float*)sPraw;                 // [128][LDO_] f32
  #pragma unroll
  for (int mt = 0; mt < 4; ++mt) {
    float inv = sTot[mt * 16 + lrow];
    #pragma unroll
    for (int r = 0; r < 8; ++r)
      sPf[(wave * 16 + r + 8 * lhi) * LDO_ + mt * 16 + lrow] = O[mt][r] * inv;
  }
  __syncthreads();
  for (int i = threadIdx.x; i < 128 * 16; i += 256) {
    int row = i >> 4, seg = i & 15;           // 16B chunks of each 64-f32 row
    copy16_l2g(xr + ((size_t)b * D_ + row) * N_ + m0 + seg * 4,
               &sPf[row * LDO_ + seg * 4]);
  }
  // async stores (if used) are drained by S_ENDPGM's implicit wait-idle
}

// ---------------------------------------------------------------------------
// Kernel 4: y = Wt·(x - x_r) + bt, BN(inference), residual ReLU.
// x_r aliased into d_out; each block reads its tile into LDS before writing.
// ---------------------------------------------------------------------------
__global__ __launch_bounds__(256)
void output_kernel(const float* __restrict__ x, const float* xr,
                   const float* __restrict__ Wt, const float* __restrict__ bt,
                   const float* __restrict__ gamma, const float* __restrict__ beta,
                   const float* __restrict__ rmean, const float* __restrict__ rvar,
                   float* out) {
  const int b  = blockIdx.x / (N_ / 16);
  const int n0 = (blockIdx.x % (N_ / 16)) * 16;
  const int tid = threadIdx.x;
  __shared__ float sxd[D_][17];
  for (int i = tid; i < D_ * 16; i += 256) {
    int d = i >> 4, n = i & 15;
    size_t gi = ((size_t)b * C_ + d) * N_ + n0 + n;
    sxd[d][n] = x[gi] - xr[gi];
  }
  __syncthreads();
  #pragma unroll 1
  for (int j = 0; j < 8; ++j) {
    int idx = tid * 8 + j;
    int c = idx >> 4, n = idx & 15;
    float acc = bt[c];
    const float* wr = Wt + c * D_;
    #pragma unroll 4
    for (int d = 0; d < D_; ++d) acc = fmaf(wr[d], sxd[d][n], acc);
    float ybn = gamma[c] * (acc - rmean[c]) * rsqrtf(rvar[c] + BN_EPS) + beta[c];
    size_t gi = ((size_t)b * C_ + c) * N_ + n0 + n;
    out[gi] = x[gi] + fmaxf(ybn, 0.f);
  }
}

// ---------------------------------------------------------------------------
extern "C" void kernel_launch(void* const* d_in, const int* in_sizes, int n_in,
                              void* d_out, int out_size, void* d_ws, size_t ws_size,
                              hipStream_t stream) {
  const float* x     = (const float*)d_in[0];
  const float* Wqk   = (const float*)d_in[1];
  const float* Wv    = (const float*)d_in[2];
  const float* bv    = (const float*)d_in[3];
  const float* Wt    = (const float*)d_in[4];
  const float* bt    = (const float*)d_in[5];
  const float* gamma = (const float*)d_in[6];
  const float* beta  = (const float*)d_in[7];
  const float* rmean = (const float*)d_in[8];
  const float* rvar  = (const float*)d_in[9];
  float* out = (float*)d_out;

  char* w = (char*)d_ws;
  size_t off = 0;
  auto take = [&](size_t bytes) {
    char* p = w + off;
    off = (off + bytes + 255) & ~(size_t)255;
    return p;
  };
  __bf16* qb   = (__bf16*)take((size_t)B_ * N_ * D_ * 2);  // 4 MB
  __bf16* vb   = (__bf16*)take((size_t)B_ * D_ * N_ * 2);  // 4 MB
  float*  rmax = (float*)take((size_t)B_ * N_ * 4);        // 64 KB
  float*  rsum = (float*)take((size_t)B_ * N_ * 4);        // 64 KB
  __bf16* wqb  = (__bf16*)take((size_t)D_ * C_ * 2);       // 32 KB
  __bf16* wvb  = (__bf16*)take((size_t)D_ * C_ * 2);       // 32 KB
  float*  xr   = out;  // x_r aliased into d_out (fully overwritten at the end)

  convw_kernel<<<(D_ * C_) / 256, 256, 0, stream>>>(Wqk, Wv, wqb, wvb);
  proj_kernel<<<B_ * (N_ / 128), 256, 0, stream>>>(x, wqb, wvb, bv, qb, vb);
  rowstats_kernel<<<B_ * (N_ / 128), 256, 0, stream>>>(qb, rmax, rsum);
  attend_kernel<<<B_ * (N_ / MT_), 256, 0, stream>>>(qb, vb, rmax, rsum, xr);
  output_kernel<<<B_ * (N_ / 16), 256, 0, stream>>>(x, xr, Wt, bt, gamma, beta,
                                                    rmean, rvar, out);
  (void)in_sizes; (void)n_in; (void)out_size; (void)ws_size;
}
